// CPBAttention_82085414961218
// MI455X (gfx1250) — compile-verified, matrix operand order checked by reading, not run on hardware
//
#include <hip/hip_runtime.h>

#define BB 2
#define CC 256
#define NN 4096
#define HH 32
#define HDIM 8
#define KTOP 512

typedef __attribute__((ext_vector_type(16))) __bf16 v16bf;
typedef __attribute__((ext_vector_type(8)))  float  v8f;

union FragB16 { v16bf v; unsigned u[8]; uint4 q[2]; };

__device__ __forceinline__ unsigned short bf16b(float f) {
  unsigned x = __float_as_uint(f);
  unsigned r = x + 0x7FFFu + ((x >> 16) & 1u);
  return (unsigned short)(r >> 16);
}

// ---------------- channel mean/max ----------------
__global__ void k_reduce(const float* __restrict__ x, float* __restrict__ am) {
  int id = blockIdx.x * blockDim.x + threadIdx.x;
  if (id >= BB * NN) return;
  int b = id / NN, n = id % NN;
  const float* p = x + (size_t)b * CC * NN + n;
  float s = 0.f, m = -3.4e38f;
  for (int c = 0; c < CC; ++c) { float v = p[(size_t)c * NN]; s += v; m = fmaxf(m, v); }
  am[(b * 2 + 0) * NN + n] = s * (1.0f / CC);
  am[(b * 2 + 1) * NN + n] = m;
}

// ---------------- 7x7x7 conv + sigmoid ----------------
__global__ void k_spa(const float* __restrict__ am, const float* __restrict__ w,
                      float* __restrict__ scores) {
  int id = blockIdx.x * blockDim.x + threadIdx.x;
  if (id >= BB * NN) return;
  int b = id / NN, n = id % NN;
  int d = n >> 8, h = (n >> 4) & 15, wc = n & 15;
  float s = 0.f;
  for (int ic = 0; ic < 2; ++ic) {
    const float* vol = am + (b * 2 + ic) * NN;
    const float* wt  = w + ic * 343;
    for (int kd = 0; kd < 7; ++kd) { int zd = d + kd - 3; if ((unsigned)zd >= 16u) continue;
      for (int kh = 0; kh < 7; ++kh) { int zh = h + kh - 3; if ((unsigned)zh >= 16u) continue;
        for (int kw = 0; kw < 7; ++kw) { int zw = wc + kw - 3; if ((unsigned)zw >= 16u) continue;
          s += vol[(zd << 8) + (zh << 4) + zw] * wt[kd * 49 + kh * 7 + kw];
        } } }
  }
  scores[id] = 1.0f / (1.0f + __expf(-s));
}

// ---------------- per-batch top-512 via bitonic sort in LDS ----------------
__global__ void k_topk(const float* __restrict__ scores, int* __restrict__ idx) {
  __shared__ float key[NN];
  __shared__ int   val[NN];
  int b = blockIdx.x, t = threadIdx.x;
  for (int i = t; i < NN; i += 1024) { key[i] = scores[b * NN + i]; val[i] = i; }
  __syncthreads();
  for (int k = 2; k <= NN; k <<= 1) {
    for (int j = k >> 1; j > 0; j >>= 1) {
      for (int i = t; i < NN; i += 1024) {
        int ixj = i ^ j;
        if (ixj > i) {
          bool asc = (i & k) != 0;   // top-level descending
          if ((key[i] > key[ixj]) == asc) {
            float tk = key[i]; key[i] = key[ixj]; key[ixj] = tk;
            int   tv = val[i]; val[i] = val[ixj]; val[ixj] = tv;
          }
        }
      }
      __syncthreads();
    }
  }
  for (int i = t; i < KTOP; i += 1024) idx[b * KTOP + i] = val[i];
}

// ---------------- weight prep: bf16, transposed so k is contiguous per out-col ----
__global__ void k_wprep(const float* __restrict__ W, unsigned short* __restrict__ Wt,
                        int Nout, int wt) {
  int id = blockIdx.x * blockDim.x + threadIdx.x;
  if (id >= Nout * 256) return;
  int n = id >> 8, k = id & 255;
  float v = wt ? W[(size_t)n * 256 + k] : W[(size_t)k * Nout + n];
  Wt[id] = bf16b(v);
}

// ---------------- generic WMMA GEMM: rows x 256 @ 256 x Nout ----------------
// MODE 0: A = x (B,C,N) f32 -> out0 = k bf16 (rows,256), out1 = v f32 (B,C,N)
// MODE 1: A = x (B,C,N) f32 -> out0 = q bf16 (rows,256), scaled by 1/sqrt(hd)
// MODE 2: A = bf16 (rows,256) -> out0 = d_out f32 (B,C,N) (store)
// MODE 3: A = bf16 (rows,256) -> out0 = d_out f32 (B,C,N) (accumulate)
template <int MODE>
__global__ void __launch_bounds__(128)
k_gemm(const void* __restrict__ Asrc, const unsigned short* __restrict__ Wt,
       const float* __restrict__ bias, void* __restrict__ out0, void* __restrict__ out1) {
  __shared__ __align__(16) unsigned short Abuf[16 * 256];
  int row0 = blockIdx.x * 16;
  int b = row0 / NN, n0 = row0 % NN;
  int t = threadIdx.x;
  if (MODE <= 1) {
    const float* x = (const float*)Asrc;          // (B,C,N): transpose-stage via LDS
    #pragma unroll
    for (int rep = 0; rep < 2; ++rep) {
      int k = t + rep * 128;
      const float* src = x + ((size_t)(b * CC + k)) * NN + n0;
      #pragma unroll
      for (int j = 0; j < 16; ++j) Abuf[j * 256 + k] = bf16b(src[j]);
    }
  } else {
    const unsigned short* A = (const unsigned short*)Asrc;  // (rows,256) bf16
    int m = t >> 3, ch = (t & 7) * 32;
    const uint4* src = (const uint4*)(A + (size_t)(row0 + m) * 256 + ch);
    uint4* dst = (uint4*)&Abuf[m * 256 + ch];
    #pragma unroll
    for (int i = 0; i < 4; ++i) dst[i] = src[i];
  }
  __syncthreads();

  int lane = t & 31, wave = t >> 5;
  int nout0 = blockIdx.y * 64 + wave * 16;
  int ncol = lane & 15, half = (lane >> 4) & 1;
  int oc = nout0 + ncol;
  v8f acc;
  { float bv = bias[oc]; for (int i = 0; i < 8; ++i) acc[i] = bv; }

  const unsigned* Ab = (const unsigned*)Abuf;               // 16 rows x 128 dwords
  const unsigned* Wb = (const unsigned*)Wt + (size_t)oc * 128;

  #pragma unroll
  for (int k0 = 0; k0 < 256; k0 += 32) {
    int ko = (k0 >> 1) + half * 4;
    __builtin_prefetch(&Wb[ko], 0, 1);
    FragB16 fa, fb;
    fa.q[0] = *(const uint4*)&Ab[ncol * 128 + ko];
    fa.q[1] = *(const uint4*)&Ab[ncol * 128 + ko + 8];
    fb.q[0] = *(const uint4*)&Wb[ko];
    fb.q[1] = *(const uint4*)&Wb[ko + 8];
    acc = __builtin_amdgcn_wmma_f32_16x16x32_bf16(false, fa.v, false, fb.v,
                                                  (short)0, acc, false, false);
  }

  #pragma unroll
  for (int v = 0; v < 8; ++v) {
    int m = v + half * 8;
    float val = acc[v];
    if (MODE == 0) {
      if (oc < CC) ((unsigned short*)out0)[((size_t)(row0 + m)) * CC + oc] = bf16b(val);
      else         ((float*)out1)[((size_t)(b * CC + (oc - CC))) * NN + n0 + m] = val;
    } else if (MODE == 1) {
      // fold attention scale 1/sqrt(8) into q
      ((unsigned short*)out0)[((size_t)(row0 + m)) * CC + oc] =
          bf16b(val * 0.35355339059327373f);
    } else if (MODE == 2) {
      ((float*)out0)[((size_t)(b * CC + oc)) * NN + n0 + m] = val;
    } else {
      float* o = (float*)out0 + ((size_t)(b * CC + oc)) * NN + n0 + m;
      *o += val;
    }
  }
}

// ---------------- gather top-k K / V rows ----------------
__global__ void k_gather(const unsigned short* __restrict__ kbf, const float* __restrict__ vvol,
                         const int* __restrict__ idx, unsigned short* __restrict__ kg,
                         unsigned short* __restrict__ vg) {
  int id = blockIdx.x * blockDim.x + threadIdx.x;
  if (id >= BB * KTOP * CC) return;
  int c = id & 255, k = (id >> 8) & 511, b = id >> 17;
  int src = idx[b * KTOP + k];
  kg[id] = kbf[((size_t)(b * NN + src)) * CC + c];
  vg[id] = bf16b(vvol[((size_t)(b * CC + c)) * NN + src]);
}

// ---------------- fused attention: one wave per (b, head, 16 q-rows) ----------------
__global__ void __launch_bounds__(32)
k_attn(const unsigned short* __restrict__ kg, const unsigned short* __restrict__ vg,
       const unsigned short* __restrict__ qb, unsigned short* __restrict__ obuf) {
  // K stored in padded fragment rows [key][32] (d 0..7 data, 8..31 zero) so the
  // B-fragment is two unconditional ds_load_b128 for every lane (no selects).
  __shared__ __align__(16) unsigned short Kb[KTOP * 32];
  __shared__ __align__(16) unsigned short Vbt[16 * KTOP];   // [d][key], rows 8..15 never written
  __shared__ __align__(16) unsigned short Qb[16 * 32];      // zero-padded K->32
  __shared__ __align__(16) float          Sb[16 * KTOP];
  __shared__ __align__(16) unsigned short Pb[16 * KTOP];
  int id = blockIdx.x;
  int nb = id & 255, h = (id >> 8) & 31, b = id >> 13;
  int lane = threadIdx.x;
  uint4 z4 = make_uint4(0u, 0u, 0u, 0u);

  for (int k = lane; k < KTOP; k += 32) {
    uint4* kd = (uint4*)&Kb[k * 32];
    kd[0] = *(const uint4*)(kg + ((size_t)(b * KTOP + k)) * CC + h * HDIM);
    kd[1] = z4; kd[2] = z4; kd[3] = z4;
    uint4 vv = *(const uint4*)(vg + ((size_t)(b * KTOP + k)) * CC + h * HDIM);
    const unsigned short* vs = (const unsigned short*)&vv;
    #pragma unroll
    for (int d = 0; d < HDIM; ++d) Vbt[d * KTOP + k] = vs[d];
  }
  {   // zero Vbt rows 4..7 (uint4 256..511)
    uint4* vz = (uint4*)Vbt;
    for (int i = 256 + lane; i < 512; i += 32) vz[i] = z4;
  }
  if (lane < 16) {
    uint4* qd = (uint4*)&Qb[lane * 32];
    qd[0] = *(const uint4*)(qb + ((size_t)(b * NN + nb * 16 + lane)) * CC + h * HDIM);
    qd[1] = z4; qd[2] = z4; qd[3] = z4;
  }
  __syncthreads();

  int m = lane & 15, half = (lane >> 4) & 1;
  const unsigned* Qu = (const unsigned*)Qb;   // 16 dwords per row
  const unsigned* Ku = (const unsigned*)Kb;   // 16 dwords per key (padded)
  const unsigned* Vu = (const unsigned*)Vbt;  // 256 dwords per d-row
  const unsigned* Pu = (const unsigned*)Pb;   // 256 dwords per q-row

  // Q fragment is invariant across key tiles: hoist it.
  FragB16 fq;
  fq.q[0] = *(const uint4*)&Qu[m * 16 + half * 4];
  fq.q[1] = *(const uint4*)&Qu[m * 16 + 8 + half * 4];

  // scores: Q (16x32 padded) x K^T, 32 tiles of 16 keys (scale pre-folded into q)
  #pragma unroll
  for (int kt = 0; kt < 32; ++kt) {
    FragB16 fb;
    int key = kt * 16 + m;
    fb.q[0] = *(const uint4*)&Ku[key * 16 + half * 4];
    fb.q[1] = *(const uint4*)&Ku[key * 16 + 8 + half * 4];
    v8f c = {};
    c = __builtin_amdgcn_wmma_f32_16x16x32_bf16(false, fq.v, false, fb.v,
                                                (short)0, c, false, false);
    #pragma unroll
    for (int v = 0; v < 8; ++v)
      Sb[(v + half * 8) * KTOP + kt * 16 + m] = c[v];
  }
  __syncthreads();

  // softmax along 512: 2 lanes per row (lane&15 = row, half = column half)
  {
    float* row = Sb + m * KTOP;
    int j0 = half * 256;
    float mx = -3.4e38f;
    for (int j = 0; j < 256; ++j) mx = fmaxf(mx, row[j0 + j]);
    mx = fmaxf(mx, __shfl_xor(mx, 16));
    float s = 0.f;
    for (int j = 0; j < 256; ++j) { float e = __expf(row[j0 + j] - mx); row[j0 + j] = e; s += e; }
    s += __shfl_xor(s, 16);
    float inv = 1.0f / s;
    unsigned short* prow = Pb + m * KTOP;
    for (int j = 0; j < 256; ++j) prow[j0 + j] = bf16b(row[j0 + j] * inv);
  }
  __syncthreads();

  // out = P (16x512) x V (512x8, N padded to 16 in LDS)
  v8f acc = {};
  #pragma unroll
  for (int kt2 = 0; kt2 < 16; ++kt2) {
    FragB16 fa, fb;
    int ko = kt2 * 16 + half * 4;
    fa.q[0] = *(const uint4*)&Pu[m * 256 + ko];
    fa.q[1] = *(const uint4*)&Pu[m * 256 + ko + 8];
    fb.q[0] = *(const uint4*)&Vu[m * 256 + ko];
    fb.q[1] = *(const uint4*)&Vu[m * 256 + ko + 8];
    acc = __builtin_amdgcn_wmma_f32_16x16x32_bf16(false, fa.v, false, fb.v,
                                                  (short)0, acc, false, false);
  }
  if (m < HDIM) {
    #pragma unroll
    for (int v = 0; v < 8; ++v) {
      int row = v + half * 8;
      obuf[((size_t)(b * NN + nb * 16 + row)) * CC + h * HDIM + m] = bf16b(acc[v]);
    }
  }
}

// ---------------- depthwise 3x3x3 ----------------
__global__ void k_dw(const float* __restrict__ vvol, const float* __restrict__ wdw,
                     const float* __restrict__ bdw, unsigned short* __restrict__ dwt) {
  int id = blockIdx.x * blockDim.x + threadIdx.x;
  if (id >= BB * CC * NN) return;
  int n = id & 4095, c = (id >> 12) & 255, b = id >> 20;
  int d = n >> 8, h = (n >> 4) & 15, w = n & 15;
  const float* vol = vvol + ((size_t)(b * CC + c)) * NN;
  const float* wt = wdw + c * 27;
  float s = 0.f;
  for (int dz = -1; dz <= 1; ++dz) { int zd = d + dz; if ((unsigned)zd >= 16u) continue;
    for (int dy = -1; dy <= 1; ++dy) { int zh = h + dy; if ((unsigned)zh >= 16u) continue;
      for (int dx = -1; dx <= 1; ++dx) { int zw = w + dx; if ((unsigned)zw >= 16u) continue;
        s += vol[(zd << 8) + (zh << 4) + zw] * wt[(dz + 1) * 9 + (dy + 1) * 3 + (dx + 1)];
      } } }
  dwt[((size_t)(b * NN + n)) * CC + c] = bf16b(s + bdw[c]);
}

extern "C" void kernel_launch(void* const* d_in, const int* in_sizes, int n_in,
                              void* d_out, int out_size, void* d_ws, size_t ws_size,
                              hipStream_t stream) {
  const float* x_kv   = (const float*)d_in[0];
  const float* x_q    = (const float*)d_in[1];
  const float* w_spa  = (const float*)d_in[2];
  const float* w_kv   = (const float*)d_in[3];
  const float* b_kv   = (const float*)d_in[4];
  const float* w_q    = (const float*)d_in[5];
  const float* b_q    = (const float*)d_in[6];
  const float* w_proj = (const float*)d_in[7];
  const float* b_proj = (const float*)d_in[8];
  const float* w_dw   = (const float*)d_in[9];
  const float* b_dw   = (const float*)d_in[10];
  const float* w_pw   = (const float*)d_in[11];
  const float* b_pw   = (const float*)d_in[12];
  float* out = (float*)d_out;

  char* p = (char*)d_ws;
  auto alloc = [&](size_t bytes) { void* r = (void*)p; p += (bytes + 255) & ~(size_t)255; return r; };
  float*          am     = (float*)alloc((size_t)BB * 2 * NN * 4);
  float*          scores = (float*)alloc((size_t)BB * NN * 4);
  int*            idx    = (int*)  alloc((size_t)BB * KTOP * 4);
  unsigned short* kbf    = (unsigned short*)alloc((size_t)BB * NN * CC * 2);
  float*          vvol   = (float*)alloc((size_t)BB * CC * NN * 4);
  unsigned short* qbf    = (unsigned short*)alloc((size_t)BB * NN * CC * 2);
  unsigned short* kg     = (unsigned short*)alloc((size_t)BB * KTOP * CC * 2);
  unsigned short* vg     = (unsigned short*)alloc((size_t)BB * KTOP * CC * 2);
  unsigned short* obuf   = (unsigned short*)alloc((size_t)BB * NN * CC * 2);
  unsigned short* dwt    = (unsigned short*)alloc((size_t)BB * NN * CC * 2);
  unsigned short* wkvT   = (unsigned short*)alloc((size_t)512 * 256 * 2);
  unsigned short* wqT    = (unsigned short*)alloc((size_t)256 * 256 * 2);
  unsigned short* wprT   = (unsigned short*)alloc((size_t)256 * 256 * 2);
  unsigned short* wpwT   = (unsigned short*)alloc((size_t)256 * 256 * 2);

  // weight prep (independent of data path)
  k_wprep<<<(512 * 256 + 255) / 256, 256, 0, stream>>>(w_kv,   wkvT, 512, 0);
  k_wprep<<<(256 * 256 + 255) / 256, 256, 0, stream>>>(w_q,    wqT,  256, 0);
  k_wprep<<<(256 * 256 + 255) / 256, 256, 0, stream>>>(w_proj, wprT, 256, 0);
  k_wprep<<<(256 * 256 + 255) / 256, 256, 0, stream>>>(w_pw,   wpwT, 256, 1);

  k_reduce<<<(BB * NN + 255) / 256, 256, 0, stream>>>(x_kv, am);
  k_spa<<<(BB * NN + 127) / 128, 128, 0, stream>>>(am, w_spa, scores);
  k_topk<<<BB, 1024, 0, stream>>>(scores, idx);

  dim3 g0((BB * NN) / 16, 512 / 64);
  k_gemm<0><<<g0, 128, 0, stream>>>(x_kv, wkvT, b_kv, kbf, vvol);
  dim3 g1((BB * NN) / 16, 256 / 64);
  k_gemm<1><<<g1, 128, 0, stream>>>(x_q, wqT, b_q, qbf, nullptr);

  k_gather<<<(BB * KTOP * CC + 255) / 256, 256, 0, stream>>>(kbf, vvol, idx, kg, vg);
  k_attn<<<BB * HH * (NN / 16), 32, 0, stream>>>(kg, vg, qbf, obuf);

  k_gemm<2><<<g1, 128, 0, stream>>>(obuf, wprT, b_proj, out, nullptr);
  k_dw<<<(BB * CC * NN + 255) / 256, 256, 0, stream>>>(vvol, w_dw, b_dw, dwt);
  k_gemm<3><<<g1, 128, 0, stream>>>(dwt, wpwT, b_pw, out, nullptr);
}
